// SmoothAttention_43224550867507
// MI455X (gfx1250) — matrix operand order repaired and checked by reading, not compiled, not run on hardware
//
#include <hip/hip_runtime.h>
#include <cstddef>

typedef __attribute__((ext_vector_type(16))) _Float16 v16h;
typedef __attribute__((ext_vector_type(8)))  float    v8f;
typedef __attribute__((ext_vector_type(4)))  unsigned int u32x4;
typedef __attribute__((ext_vector_type(8)))  int i32x8;
typedef __attribute__((ext_vector_type(4)))  int i32x4;

namespace {
constexpr int   kB = 2, kC = 64, kH = 64, kW = 64, kN = 4096;
constexpr float kThresh = 0.4f;
}

// ---------------- WMMA fragment helpers (wave32, contiguous-K order, same for A and B) ---------

// A (16x32 f16, row-major tile in LDS): lane = M row; lanes 0-15 hold
// K = 0..15, lanes 16-31 hold K = 16..31.
__device__ __forceinline__ v16h frag_a_rowmajor(const _Float16* s, int ld) {
  const int lane = threadIdx.x & 31;
  const int row  = lane & 15;
  const int kb   = (lane < 16) ? 0 : 16;
  const _Float16* p = s + row * ld;
  v16h a;
#pragma unroll
  for (int i = 0; i < 16; ++i) a[i] = p[kb + i];
  return a;
}

// B (32x16 f16, [K][N] row-major tile in LDS): lane = N column;
// lanes 0-15 hold K = 0..15, lanes 16-31 hold K = 16..31.
__device__ __forceinline__ v16h frag_b_rowmajor(const _Float16* s, int ld) {
  const int lane = threadIdx.x & 31;
  const int col  = lane & 15;
  const int kb   = (lane < 16) ? 0 : 16;
  v16h b;
#pragma unroll
  for (int i = 0; i < 16; ++i) b[i] = s[(kb + i) * ld + col];
  return b;
}

// B fragment when B is stored transposed in LDS as s[n][k] (contiguous K per
// lane -> conflict-free, vectorizable LDS reads).
__device__ __forceinline__ v16h frag_b_nk(const _Float16* s, int ld) {
  const int lane = threadIdx.x & 31;
  const _Float16* p = s + (size_t)(lane & 15) * ld + ((lane < 16) ? 0 : 16);
  v16h b;
#pragma unroll
  for (int i = 0; i < 16; ++i) b[i] = p[i];
  return b;
}

__device__ __forceinline__ v8f wmma_f32_f16(v16h a, v16h b, v8f c) {
  return __builtin_amdgcn_wmma_f32_16x16x32_f16(false, a, false, b,
                                                (short)0, c, false, false);
}

__device__ __forceinline__ float wave_max(float v) {
#pragma unroll
  for (int o = 16; o > 0; o >>= 1) v = fmaxf(v, __shfl_xor(v, o, 32));
  return v;
}
__device__ __forceinline__ float wave_sum(float v) {
#pragma unroll
  for (int o = 16; o > 0; o >>= 1) v += __shfl_xor(v, o, 32);
  return v;
}

// ------------- Tensor Data Mover: 2-D tile load, global -> LDS --------------
// D# per CDNA5 ISA ch.8: group0 = {count, lds_addr, global_addr, type=2},
// group1 = {data_size=2B, tensor dims == tile dims (tile fully in-bounds),
// tile_dim0/1, tensor_dim0_stride}. Groups 2/3 zero (2-D tensor).
// Issued by one wave; completion via TENSORcnt, published via barrier.
#define RFL(x) __builtin_amdgcn_readfirstlane((int)(x))

__device__ __forceinline__ void tdm_load_2d_f16(unsigned lds_addr,
                                                const void* gptr,
                                                unsigned tile_w,   // elements
                                                unsigned tile_h,   // rows
                                                unsigned stride) { // elements
  const unsigned long long ga = (unsigned long long)(size_t)gptr;
  u32x4 g0;
  g0[0] = (unsigned)RFL(1u);                                  // count=1 (valid)
  g0[1] = (unsigned)RFL(lds_addr);                            // lds_addr
  g0[2] = (unsigned)RFL((unsigned)(ga & 0xffffffffu));        // gaddr[31:0]
  g0[3] = (unsigned)RFL(((unsigned)(ga >> 32) & 0x01ffffffu) |
                        (2u << 30));                          // gaddr[56:32]|type=2
  i32x8 g1;
  g1[0] = RFL(1u << 16);                       // data_size=1 (2 bytes), mask=0
  g1[1] = RFL((tile_w & 0xffffu) << 16);       // tensor_dim0[15:0]
  g1[2] = RFL(((tile_w >> 16) & 0xffffu) | ((tile_h & 0xffffu) << 16));
  g1[3] = RFL(((tile_h >> 16) & 0xffffu) | ((tile_w & 0xffffu) << 16)); // tile_dim0
  g1[4] = RFL(tile_h & 0xffffu);               // tile_dim1, tile_dim2=0
  g1[5] = RFL(stride);                         // tensor_dim0_stride[31:0]
  g1[6] = RFL(0);
  g1[7] = RFL(0);
  const i32x4 z4 = {0, 0, 0, 0};
  const i32x8 z8 = {0, 0, 0, 0, 0, 0, 0, 0};
  // amdgpu-toolchain (clang-23) 6-arg form: (g0, g1, g2, g3, g4, cpol)
  __builtin_amdgcn_tensor_load_to_lds(g0, g1, z4, z4, z8, 0);
}

// -------------------- Kernel 1: q/k/v = W @ x + b  (f16 out) -----------------
extern "C" __global__ void __launch_bounds__(256)
qkv_proj_kernel(const float* __restrict__ x,
                const float* __restrict__ Wq, const float* __restrict__ bq,
                const float* __restrict__ Wk, const float* __restrict__ bk,
                const float* __restrict__ Wv, const float* __restrict__ bv,
                _Float16* __restrict__ q16, _Float16* __restrict__ k16,
                _Float16* __restrict__ v16) {
  extern __shared__ unsigned char smem_raw[];
  _Float16* xs = (_Float16*)smem_raw;   // [64][64]  x tile (c, n)
  _Float16* wm = xs + 64 * 64;          // [3][64][64] Wq,Wk,Wv (o, c)
  const int t  = threadIdx.x;
  const int b  = blockIdx.x >> 6;
  const int n0 = (blockIdx.x & 63) * 64;

  for (int idx = t; idx < 64 * 64; idx += 256) {
    int c = idx >> 6, j = idx & 63;
    xs[idx]        = (_Float16)x[((size_t)b * kC + c) * kN + n0 + j];
    wm[idx]        = (_Float16)Wq[idx];
    wm[4096 + idx] = (_Float16)Wk[idx];
    wm[8192 + idx] = (_Float16)Wv[idx];
  }
  __syncthreads();

  const int wave = t >> 5, lane = t & 31;
  const int col = lane & 15, mo = (lane >> 4) << 3;
  for (int tt = wave; tt < 48; tt += 8) {   // 3 matrices x 16 (4Mx4N) tiles
    const int mat = tt >> 4, sub = tt & 15, mt = sub >> 2, nt = sub & 3;
    v8f acc = {};
#pragma unroll
    for (int ks = 0; ks < 64; ks += 32) {
      v16h a  = frag_a_rowmajor(wm + mat * 4096 + mt * 16 * 64 + ks, 64);
      v16h bf = frag_b_rowmajor(xs + ks * 64 + nt * 16, 64);
      acc = wmma_f32_f16(a, bf, acc);
    }
    const float* bias = (mat == 0) ? bq : (mat == 1) ? bk : bv;
    _Float16*    dst  = (mat == 0) ? q16 : (mat == 1) ? k16 : v16;
#pragma unroll
    for (int r = 0; r < 8; ++r) {
      int row = mt * 16 + mo + r;
      dst[((size_t)b * kC + row) * kN + n0 + nt * 16 + col] =
          (_Float16)(acc[r] + bias[row]);
    }
  }
}

// -------- Kernel 2: energy (q^T k) + fused softmax, 16 rows per WG ----------
// 16x4096 f32 energy strip lives in LDS (256 KiB of CDNA5's 320 KiB/WGP);
// k-tiles are double-buffered and fetched by the Tensor Data Mover so the
// DMA of tile j+1 overlaps the WMMAs on tile j.
extern "C" __global__ void __launch_bounds__(256)
energy_softmax_kernel(const _Float16* __restrict__ q16,
                      const _Float16* __restrict__ k16,
                      _Float16* __restrict__ attn) {
  extern __shared__ unsigned char smem_raw[];
  float*    erow = (float*)smem_raw;             // [16][4096] energy strip
  _Float16* qs   = (_Float16*)(erow + 16 * kN);  // [16][64]   q^T tile (i, c)
  _Float16* ks   = qs + 16 * 64;                 // [2][64][128] k tiles (c, j)
  const int t = threadIdx.x, wave = t >> 5, lane = t & 31;
  const int b  = blockIdx.x >> 8;
  const int i0 = (blockIdx.x & 255) * 16;
  const unsigned lds_ks0 = (unsigned)(size_t)ks;
  const _Float16* kbase = k16 + (size_t)b * kC * kN;

  for (int idx = t; idx < 16 * 64; idx += 256) {
    int i = idx >> 6, c = idx & 63;
    qs[idx] = q16[((size_t)b * kC + c) * kN + i0 + i];
  }

  constexpr int kJIters = kN / 128;
  if (wave == 0)  // prologue: DMA tile 0
    tdm_load_2d_f16(lds_ks0, kbase, 128, 64, kN);

  for (int jt = 0; jt < kJIters; ++jt) {
    const int j0 = jt * 128;
    if (wave == 0) {
      if (jt + 1 < kJIters) {
        tdm_load_2d_f16(lds_ks0 + ((jt + 1) & 1) * (64 * 128 * 2),
                        kbase + (size_t)(j0 + 128), 128, 64, kN);
        __builtin_amdgcn_s_wait_tensorcnt(1);  // tile jt landed
      } else {
        __builtin_amdgcn_s_wait_tensorcnt(0);
      }
    }
    __syncthreads();  // publish tile jt to all waves

    const _Float16* kt = ks + (jt & 1) * (64 * 128);
    v8f acc = {};
#pragma unroll
    for (int ks0 = 0; ks0 < 64; ks0 += 32) {
      v16h a  = frag_a_rowmajor(qs + ks0, 64);
      v16h bf = frag_b_rowmajor(kt + ks0 * 128 + wave * 16, 128);
      acc = wmma_f32_f16(a, bf, acc);
    }
    float* dst = erow + j0 + wave * 16;   // wave w owns cols j0 + 16w..+15
    const int col = lane & 15, mo = (lane >> 4) << 3;
#pragma unroll
    for (int r = 0; r < 8; ++r) dst[(size_t)(r + mo) * kN + col] = acc[r];
    __syncthreads();  // tile jt fully consumed before DMA overwrites it
  }

  // softmax: wave w owns rows 2w and 2w+1
#pragma unroll
  for (int rr = 0; rr < 2; ++rr) {
    const int row = wave * 2 + rr;
    float* e = erow + (size_t)row * kN;
    float mx = -3.4e38f;
    for (int m = lane; m < kN; m += 32) mx = fmaxf(mx, e[m]);
    mx = wave_max(mx);
    float s = 0.f;
    for (int m = lane; m < kN; m += 32) {
      float v = __expf(e[m] - mx);
      e[m] = v;
      s += v;
    }
    s = wave_sum(s);
    const float inv = 1.f / s;
    _Float16* dst = attn + ((size_t)b * kN + i0 + row) * kN;
    for (int m = lane; m < kN; m += 32) dst[m] = (_Float16)(e[m] * inv);
  }
}

// ------ Kernel 3: Chebyshev-gated 8-neighbor smoothing, one row per WG ------
extern "C" __global__ void __launch_bounds__(256)
smooth_kernel(const _Float16* __restrict__ attn, _Float16* __restrict__ fin) {
  extern __shared__ unsigned char smem_raw[];
  float* centers = (float*)smem_raw;  // [4096]
  float* sums    = centers + kN;      // [4096]
  float* red     = sums + kN;         // [8]
  const int t = threadIdx.x, wave = t >> 5, lane = t & 31;
  const int b = blockIdx.x >> 12;
  const int r = blockIdx.x & (kN - 1);
  const int h = r >> 6, w = r & 63;

  int nb[8];
  {
    int q = 0;
#pragma unroll
    for (int di = -1; di <= 1; ++di)
#pragma unroll
      for (int dj = -1; dj <= 1; ++dj) {
        if (di == 0 && dj == 0) continue;
        int hh = min(max(h + di, 0), kH - 1);  // 'edge' padding == clamp
        int ww = min(max(w + dj, 0), kW - 1);
        nb[q++] = hh * kW + ww;
      }
  }

  const _Float16* A = attn + (size_t)b * kN * kN;
  float mymax = 0.f;
  for (int m = t; m < kN; m += 256) {
    __builtin_prefetch(&A[(size_t)r * kN + m + 256], 0, 1);
    const float cv = (float)A[(size_t)r * kN + m];
    centers[m] = cv;
    float s = 0.f;
#pragma unroll
    for (int i = 0; i < 8; ++i) {
      const float v = (float)A[(size_t)nb[i] * kN + m];
      s += v;
      mymax = fmaxf(mymax, fabsf(v - cv));  // max over (nbr, m) == max of max
    }
    sums[m] = s;
  }
  mymax = wave_max(mymax);
  if (lane == 0) red[wave] = mymax;
  __syncthreads();
  float cheb = red[0];
#pragma unroll
  for (int i = 1; i < 8; ++i) cheb = fmaxf(cheb, red[i]);
  const bool sm = cheb > kThresh;
  _Float16* dst = fin + ((size_t)b * kN + r) * kN;
  for (int m = t; m < kN; m += 256)
    dst[m] = (_Float16)(sm ? sums[m] * 0.125f : centers[m]);
}

// ---------- Kernel 4: out = gamma * (v @ final^T) + x  (fp32 out) -----------
// v/final tiles double-buffered in LDS, fetched by the TDM: DMA of step m+1
// overlaps the 4 WMMAs of step m.
extern "C" __global__ void __launch_bounds__(256)
out_gemm_kernel(const _Float16* __restrict__ v16,
                const _Float16* __restrict__ fin,
                const float* __restrict__ x,
                const float* __restrict__ gamma,
                float* __restrict__ out) {
  extern __shared__ unsigned char smem_raw[];
  _Float16* vs = (_Float16*)smem_raw;  // [2][64][64] v   (c, m)
  _Float16* fs = vs + 2 * 64 * 64;     // [2][64][64] fin (k, m) -> B^T in LDS
  const int t = threadIdx.x, wave = t >> 5, lane = t & 31;
  const int b  = blockIdx.x >> 6;
  const int k0 = (blockIdx.x & 63) * 64;
  const unsigned lds_vs = (unsigned)(size_t)vs;
  const unsigned lds_fs = (unsigned)(size_t)fs;
  const _Float16* vbase = v16 + (size_t)b * kC * kN;
  const _Float16* fbase = fin + ((size_t)b * kN + k0) * kN;

  v8f acc0 = {}, acc1 = {};
  const int t0 = wave, t1 = wave + 8;        // 16 output tiles / 8 waves
  const int mt0 = t0 >> 2, nt0 = t0 & 3;
  const int mt1 = t1 >> 2, nt1 = t1 & 3;

  constexpr int kMIters = kN / 64;
  if (wave == 0) {  // prologue: DMA step 0 tile pair
    tdm_load_2d_f16(lds_vs, vbase, 64, 64, kN);
    tdm_load_2d_f16(lds_fs, fbase, 64, 64, kN);
  }

  for (int mt = 0; mt < kMIters; ++mt) {
    const int m0 = mt * 64;
    if (wave == 0) {
      if (mt + 1 < kMIters) {
        const unsigned buf = ((mt + 1) & 1) * (64 * 64 * 2);
        tdm_load_2d_f16(lds_vs + buf, vbase + (size_t)(m0 + 64), 64, 64, kN);
        tdm_load_2d_f16(lds_fs + buf, fbase + (size_t)(m0 + 64), 64, 64, kN);
        __builtin_amdgcn_s_wait_tensorcnt(2);  // step mt's pair landed
      } else {
        __builtin_amdgcn_s_wait_tensorcnt(0);
      }
    }
    __syncthreads();  // publish step mt tiles

    const _Float16* vt = vs + (mt & 1) * (64 * 64);
    const _Float16* ft = fs + (mt & 1) * (64 * 64);
#pragma unroll
    for (int ks = 0; ks < 64; ks += 32) {
      v16h a0 = frag_a_rowmajor(vt + mt0 * 16 * 64 + ks, 64);
      v16h b0 = frag_b_nk(ft + nt0 * 16 * 64 + ks, 64);
      acc0 = wmma_f32_f16(a0, b0, acc0);
      v16h a1 = frag_a_rowmajor(vt + mt1 * 16 * 64 + ks, 64);
      v16h b1 = frag_b_nk(ft + nt1 * 16 * 64 + ks, 64);
      acc1 = wmma_f32_f16(a1, b1, acc1);
    }
    __syncthreads();  // consumed before DMA overwrites this buffer
  }

  const float g = gamma[0];
  const int col = lane & 15, mo = (lane >> 4) << 3;
#pragma unroll
  for (int r = 0; r < 8; ++r) {
    {
      int row = mt0 * 16 + mo + r;
      size_t oi = ((size_t)b * kC + row) * kN + k0 + nt0 * 16 + col;
      out[oi] = g * acc0[r] + x[oi];
    }
    {
      int row = mt1 * 16 + mo + r;
      size_t oi = ((size_t)b * kC + row) * kN + k0 + nt1 * 16 + col;
      out[oi] = g * acc1[r] + x[oi];
    }
  }
}

// ------------------------------- launcher -----------------------------------
extern "C" void kernel_launch(void* const* d_in, const int* in_sizes, int n_in,
                              void* d_out, int out_size, void* d_ws,
                              size_t ws_size, hipStream_t stream) {
  const float* x     = (const float*)d_in[0];
  const float* Wq    = (const float*)d_in[1];
  const float* bq    = (const float*)d_in[2];
  const float* Wk    = (const float*)d_in[3];
  const float* bk    = (const float*)d_in[4];
  const float* Wv    = (const float*)d_in[5];
  const float* bv    = (const float*)d_in[6];
  const float* gamma = (const float*)d_in[7];
  float* out = (float*)d_out;

  // Workspace: q/k/v f16 (3 x 1 MiB) + attn f16 (64 MiB) + final f16 (64 MiB)
  unsigned char* ws = (unsigned char*)d_ws;
  const size_t szQKV = (size_t)kB * kC * kN * sizeof(_Float16);
  const size_t szAtt = (size_t)kB * kN * kN * sizeof(_Float16);
  _Float16* q16  = (_Float16*)(ws);
  _Float16* k16  = (_Float16*)(ws + szQKV);
  _Float16* v16  = (_Float16*)(ws + 2 * szQKV);
  _Float16* attn = (_Float16*)(ws + 3 * szQKV);
  _Float16* fin  = (_Float16*)(ws + 3 * szQKV + szAtt);

  // Kernel 2: 290 KiB dynamic LDS (CDNA5 allows up to 320 KiB per workgroup).
  const int k2_lds =
      16 * kN * (int)sizeof(float) + 16 * 64 * 2 + 2 * 64 * 128 * 2;
  (void)hipFuncSetAttribute((const void*)energy_softmax_kernel,
                            hipFuncAttributeMaxDynamicSharedMemorySize,
                            k2_lds);

  qkv_proj_kernel<<<kB * (kN / 64), 256, 4 * 64 * 64 * (int)sizeof(_Float16),
                    stream>>>(x, Wq, bq, Wk, bk, Wv, bv, q16, k16, v16);
  energy_softmax_kernel<<<kB * (kN / 16), 256, k2_lds, stream>>>(q16, k16,
                                                                 attn);
  smooth_kernel<<<kB * kN, 256, (2 * kN + 8) * (int)sizeof(float), stream>>>(
      attn, fin);
  out_gemm_kernel<<<kB * (kN / 64), 256,
                    4 * 64 * 64 * (int)sizeof(_Float16), stream>>>(
      v16, fin, x, gamma, out);

  (void)in_sizes; (void)n_in; (void)out_size; (void)ws_size;
}
